// GraphLayer_30605936951829
// MI455X (gfx1250) — compile-verified, matrix operand order checked by reading, not run on hardware
//
#include <hip/hip_runtime.h>

typedef __attribute__((ext_vector_type(16))) __bf16 v16bf;
typedef __attribute__((ext_vector_type(2)))  __bf16 v2bf;
typedef __attribute__((ext_vector_type(8)))  float  v8f;

#define D        128
#define NNODES   50000
#define NEDGES   640000

#if defined(__has_builtin)
#if __has_builtin(__builtin_amdgcn_cvt_pk_bf16_f32)
#define HAVE_CVT_PK_BF16 1
#endif
#endif

__device__ __forceinline__ unsigned short f2bf(float f) {
  union { float f; unsigned int u; } x; x.f = f;
  unsigned int u = x.u;
  u += 0x7fffu + ((u >> 16) & 1u);          // round-to-nearest-even
  return (unsigned short)(u >> 16);
}

union BfFrag {
  v16bf v;
  v2bf  p[8];
  unsigned short s[16];
};

// A fragment from f32 memory: two contiguous runs of 8 (run1 = run0 + 16 elements)
__device__ __forceinline__ v16bf load_a_f32(const float* run0) {
  BfFrag u;
#ifdef HAVE_CVT_PK_BF16
#pragma unroll
  for (int i = 0; i < 4; ++i) {
    u.p[i]     = __builtin_amdgcn_cvt_pk_bf16_f32(run0[2 * i],      run0[2 * i + 1]);
    u.p[4 + i] = __builtin_amdgcn_cvt_pk_bf16_f32(run0[16 + 2 * i], run0[16 + 2 * i + 1]);
  }
#else
#pragma unroll
  for (int i = 0; i < 8; ++i) { u.s[i] = f2bf(run0[i]); u.s[8 + i] = f2bf(run0[16 + i]); }
#endif
  return u.v;
}

// fragment from bf16 memory (global weights in [n][k] layout, or LDS h1 tile)
__device__ __forceinline__ v16bf load_frag_bf(const unsigned short* run0) {
  BfFrag u;
#pragma unroll
  for (int i = 0; i < 8; ++i) { u.s[i] = run0[i]; u.s[8 + i] = run0[16 + i]; }
  return u.v;
}

__device__ __forceinline__ v8f wmma_bf16(v16bf a, v16bf b, v8f c) {
  return __builtin_amdgcn_wmma_f32_16x16x32_bf16(false, a, false, b, (short)0, c, false, false);
}

// ---------------- pre-kernels ----------------

__global__ void zero_f4(float4* __restrict__ p, int n4) {
  int i = blockIdx.x * blockDim.x + threadIdx.x;
  if (i < n4) p[i] = make_float4(0.f, 0.f, 0.f, 0.f);
}

// W [K, 128] f32 row-major  ->  Wt [128][K] bf16 (dest-linear for coalesced writes)
__global__ void conv_w(const float* __restrict__ W, unsigned short* __restrict__ Wt, int K) {
  int idx = blockIdx.x * blockDim.x + threadIdx.x;
  if (idx >= K * D) return;
  int n = idx / K;
  int k = idx - n * K;
  Wt[idx] = f2bf(W[k * D + n]);
}

// ---------------- edge MLP + scatter-add (M = 32 edges per wave) ----------------

__global__ void __launch_bounds__(256) edge_kernel(
    const float* __restrict__ node_h, const float* __restrict__ edge_h,
    const int* __restrict__ src, const int* __restrict__ dst,
    const unsigned short* __restrict__ Wt1, const float* __restrict__ be1,
    const unsigned short* __restrict__ Wt2, const float* __restrict__ be2,
    float* __restrict__ agg)
{
  __shared__ unsigned short h1s[8 * 32 * D];      // 64 KB: per-wave 32x128 bf16 tile
  const int lane = threadIdx.x & 31;
  const int wave = threadIdx.x >> 5;
  const int tile = blockIdx.x * 8 + wave;         // 20000 tiles exactly
  const int row  = lane & 15;
  const int g    = lane >> 4;
  const int e0   = tile * 32;
  unsigned short* h1w = h1s + wave * (32 * D);

  // two M-subtiles of 16 edges; both share every B fragment
  const int erA = e0 + row;
  const int erB = e0 + 16 + row;
  const float* segA[3];
  segA[0] = node_h + (size_t)src[erA] * D;
  segA[1] = node_h + (size_t)dst[erA] * D;
  segA[2] = edge_h + (size_t)erA * D;
  const float* segB[3];
  segB[0] = node_h + (size_t)src[erB] * D;
  segB[1] = node_h + (size_t)dst[erB] * D;
  segB[2] = edge_h + (size_t)erB * D;

  // preload all 12 K-chunk A fragments per subtile (K = 384)
  v16bf a0[12], a1[12];
#pragma unroll
  for (int c = 0; c < 12; ++c) {
    const int off = (c & 3) * 32 + 8 * g;
    a0[c] = load_a_f32(segA[c >> 2] + off);
    a1[c] = load_a_f32(segB[c >> 2] + off);
  }

  // GEMM1: h1 = relu(e_in @ We1 + be1) -> bf16 tile in LDS
  for (int nt = 0; nt < 8; ++nt) {
    v8f acc0 = {0.f, 0.f, 0.f, 0.f, 0.f, 0.f, 0.f, 0.f};
    v8f acc1 = {0.f, 0.f, 0.f, 0.f, 0.f, 0.f, 0.f, 0.f};
#pragma unroll
    for (int c = 0; c < 12; ++c) {
      v16bf b = load_frag_bf(Wt1 + (nt * 16 + row) * 384 + c * 32 + 8 * g);
      acc0 = wmma_bf16(a0[c], b, acc0);
      acc1 = wmma_bf16(a1[c], b, acc1);
    }
    const int n = nt * 16 + row;
    const float bias = be1[n];
#pragma unroll
    for (int r = 0; r < 8; ++r) {
      float v0 = acc0[r] + bias; v0 = v0 > 0.f ? v0 : 0.f;
      float v1 = acc1[r] + bias; v1 = v1 > 0.f ? v1 : 0.f;
      h1w[(r + 8 * g) * D + n]        = f2bf(v0);
      h1w[(16 + r + 8 * g) * D + n]   = f2bf(v1);
    }
  }
  asm volatile("s_wait_dscnt 0" ::: "memory");

  // destination rows for the C/D-layout scatter
  int dr0[8], dr1[8];
#pragma unroll
  for (int r = 0; r < 8; ++r) {
    dr0[r] = dst[e0 + r + 8 * g];
    dr1[r] = dst[e0 + 16 + r + 8 * g];
  }

  // GEMM2: e_new = h1 @ We2 + be2 ; atomic scatter-add into agg[dst]
  for (int nt = 0; nt < 8; ++nt) {
    v8f acc0 = {0.f, 0.f, 0.f, 0.f, 0.f, 0.f, 0.f, 0.f};
    v8f acc1 = {0.f, 0.f, 0.f, 0.f, 0.f, 0.f, 0.f, 0.f};
#pragma unroll
    for (int c = 0; c < 4; ++c) {
      v16bf b   = load_frag_bf(Wt2 + (nt * 16 + row) * D + c * 32 + 8 * g);
      v16bf aa0 = load_frag_bf(h1w + row * D + c * 32 + 8 * g);
      v16bf aa1 = load_frag_bf(h1w + (16 + row) * D + c * 32 + 8 * g);
      acc0 = wmma_bf16(aa0, b, acc0);
      acc1 = wmma_bf16(aa1, b, acc1);
    }
    const int n = nt * 16 + row;
    const float bias = be2[n];
#pragma unroll
    for (int r = 0; r < 8; ++r) {
      unsafeAtomicAdd(agg + (size_t)dr0[r] * D + n, acc0[r] + bias);
      unsafeAtomicAdd(agg + (size_t)dr1[r] * D + n, acc1[r] + bias);
    }
  }
}

// ---------------- node MLP (M = 32 nodes per wave, tail-guarded) ----------------

__global__ void __launch_bounds__(256) node_kernel(
    const float* __restrict__ agg, const float* __restrict__ node_h,
    const unsigned short* __restrict__ Wt1, const float* __restrict__ bn1,
    const unsigned short* __restrict__ Wt2, const float* __restrict__ bn2,
    float* __restrict__ out)
{
  __shared__ unsigned short h1s[8 * 32 * D];
  const int lane = threadIdx.x & 31;
  const int wave = threadIdx.x >> 5;
  const int ntiles = (NNODES + 31) / 32;          // 1563 (last tile: 16 valid rows)
  const int tile = blockIdx.x * 8 + wave;
  if (tile >= ntiles) return;                     // wave-uniform guard, no barriers used
  const int row = lane & 15;
  const int g   = lane >> 4;
  const int n0  = tile * 32;
  unsigned short* h1w = h1s + wave * (32 * D);

  const int nA = n0 + row;                                  // always < NNODES
  const int nB = min(n0 + 16 + row, NNODES - 1);            // clamp (garbage rows unstored)
  const float* segA0 = agg    + (size_t)nA * D;
  const float* segA1 = node_h + (size_t)nA * D;
  const float* segB0 = agg    + (size_t)nB * D;
  const float* segB1 = node_h + (size_t)nB * D;

  v16bf a0[8], a1[8];                             // K = 256 -> 8 chunks
#pragma unroll
  for (int c = 0; c < 8; ++c) {
    const int off = (c & 3) * 32 + 8 * g;
    a0[c] = load_a_f32((c < 4 ? segA0 : segA1) + off);
    a1[c] = load_a_f32((c < 4 ? segB0 : segB1) + off);
  }

  for (int nt = 0; nt < 8; ++nt) {
    v8f acc0 = {0.f, 0.f, 0.f, 0.f, 0.f, 0.f, 0.f, 0.f};
    v8f acc1 = {0.f, 0.f, 0.f, 0.f, 0.f, 0.f, 0.f, 0.f};
#pragma unroll
    for (int c = 0; c < 8; ++c) {
      v16bf b = load_frag_bf(Wt1 + (nt * 16 + row) * 256 + c * 32 + 8 * g);
      acc0 = wmma_bf16(a0[c], b, acc0);
      acc1 = wmma_bf16(a1[c], b, acc1);
    }
    const int n = nt * 16 + row;
    const float bias = bn1[n];
#pragma unroll
    for (int r = 0; r < 8; ++r) {
      float v0 = acc0[r] + bias; v0 = v0 > 0.f ? v0 : 0.f;
      float v1 = acc1[r] + bias; v1 = v1 > 0.f ? v1 : 0.f;
      h1w[(r + 8 * g) * D + n]      = f2bf(v0);
      h1w[(16 + r + 8 * g) * D + n] = f2bf(v1);
    }
  }
  asm volatile("s_wait_dscnt 0" ::: "memory");

  for (int nt = 0; nt < 8; ++nt) {
    v8f acc0 = {0.f, 0.f, 0.f, 0.f, 0.f, 0.f, 0.f, 0.f};
    v8f acc1 = {0.f, 0.f, 0.f, 0.f, 0.f, 0.f, 0.f, 0.f};
#pragma unroll
    for (int c = 0; c < 4; ++c) {
      v16bf b   = load_frag_bf(Wt2 + (nt * 16 + row) * D + c * 32 + 8 * g);
      v16bf aa0 = load_frag_bf(h1w + row * D + c * 32 + 8 * g);
      v16bf aa1 = load_frag_bf(h1w + (16 + row) * D + c * 32 + 8 * g);
      acc0 = wmma_bf16(aa0, b, acc0);
      acc1 = wmma_bf16(aa1, b, acc1);
    }
    const int n = nt * 16 + row;
    const float bias = bn2[n];
#pragma unroll
    for (int r = 0; r < 8; ++r) {
      const int m0 = n0 + r + 8 * g;
      const int m1 = n0 + 16 + r + 8 * g;
      out[(size_t)m0 * D + n] = acc0[r] + bias;   // m0 always < NNODES
      if (m1 < NNODES) out[(size_t)m1 * D + n] = acc1[r] + bias;
    }
  }
}

// ---------------- launch ----------------

extern "C" void kernel_launch(void* const* d_in, const int* in_sizes, int n_in,
                              void* d_out, int out_size, void* d_ws, size_t ws_size,
                              hipStream_t stream) {
  const float* node_h = (const float*)d_in[0];
  const float* edge_h = (const float*)d_in[1];
  const int*   src    = (const int*)d_in[2];
  const int*   dst    = (const int*)d_in[3];
  const float* We1    = (const float*)d_in[4];
  const float* be1    = (const float*)d_in[5];
  const float* We2    = (const float*)d_in[6];
  const float* be2    = (const float*)d_in[7];
  const float* Wn1    = (const float*)d_in[8];
  const float* bn1    = (const float*)d_in[9];
  const float* Wn2    = (const float*)d_in[10];
  const float* bn2    = (const float*)d_in[11];
  float* out = (float*)d_out;

  char* ws = (char*)d_ws;
  float* agg = (float*)ws;                                    // [N, D] f32
  size_t aggBytes = (size_t)NNODES * D * sizeof(float);
  unsigned short* Wt_e1 = (unsigned short*)(ws + aggBytes);   // bf16 [128][384]
  unsigned short* Wt_e2 = Wt_e1 + 384 * D;                    // bf16 [128][128]
  unsigned short* Wt_n1 = Wt_e2 + D * D;                      // bf16 [128][256]
  unsigned short* Wt_n2 = Wt_n1 + 256 * D;                    // bf16 [128][128]

  const int n4 = NNODES * D / 4;
  zero_f4<<<(n4 + 255) / 256, 256, 0, stream>>>((float4*)agg, n4);
  conv_w<<<(384 * D + 255) / 256, 256, 0, stream>>>(We1, Wt_e1, 384);
  conv_w<<<(D * D + 255) / 256, 256, 0, stream>>>(We2, Wt_e2, D);
  conv_w<<<(256 * D + 255) / 256, 256, 0, stream>>>(Wn1, Wt_n1, 256);
  conv_w<<<(D * D + 255) / 256, 256, 0, stream>>>(Wn2, Wt_n2, D);

  const int edgeTiles = NEDGES / 32;                          // 20000
  edge_kernel<<<edgeTiles / 8, 256, 0, stream>>>(node_h, edge_h, src, dst,
                                                 Wt_e1, be1, Wt_e2, be2, agg);
  const int nodeTiles = (NNODES + 31) / 32;                   // 1563
  node_kernel<<<(nodeTiles + 7) / 8, 256, 0, stream>>>(agg, node_h,
                                                       Wt_n1, bn1, Wt_n2, bn2, out);
}
